// Conv3DMod_37692632990088
// MI455X (gfx1250) — matrix operand order repaired and checked by reading, not compiled, not run on hardware
//
#include <hip/hip_runtime.h>

typedef __attribute__((ext_vector_type(16))) __bf16 v16bf;
typedef __attribute__((ext_vector_type(8)))  float  v8f;

#define Bn 4
#define Cc 128
#define Oo 128
#define Tt 16
#define Hh 64
#define Wn 64
#define KTAPS 27          // 3*3*3
#define KRED (Cc * KTAPS) // 3456
#define XS 136            // padded channel stride for LDS stage (128 + 8)
#define OS 68             // padded W stride for output transpose

// ---------------------------------------------------------------------------
// Kernel 1: per-(b,o) weight modulation + demodulation, packed to bf16 in the
// layout [b][o][kt*9+kh*3+kw][i] so GEMM B-fragments read contiguous channels.
// ---------------------------------------------------------------------------
__global__ __launch_bounds__(256) void modw_kernel(
    const float* __restrict__ W,     // [O][C][3][3][3]
    const float* __restrict__ cond,  // [B][C]
    __bf16* __restrict__ wmod) {     // [B][O][27][C]
  const int bo = blockIdx.x;
  const int b  = bo >> 7;
  const int o  = bo & 127;
  const float* wp = W + (size_t)o * KRED;
  const float* cp = cond + b * Cc;

  __shared__ float red[256];
  float s = 0.f;
  for (int idx = threadIdx.x; idx < KRED; idx += 256) {
    const int i = idx / KTAPS;
    const float v = wp[idx] * (cp[i] + 1.0f);
    s += v * v;
  }
  red[threadIdx.x] = s;
  __syncthreads();
  for (int off = 128; off > 0; off >>= 1) {
    if (threadIdx.x < off) red[threadIdx.x] += red[threadIdx.x + off];
    __syncthreads();
  }
  const float scale = rsqrtf(fmaxf(red[0], 1e-8f));

  __bf16* dst = wmod + (size_t)bo * KRED;
  for (int idx = threadIdx.x; idx < KRED; idx += 256) {
    const int k = idx >> 7;   // tap index (dest-major)
    const int i = idx & 127;  // channel (dest-minor)
    const float v = wp[i * KTAPS + k] * (cp[i] + 1.0f) * scale;
    dst[idx] = (__bf16)v;
  }
}

// ---------------------------------------------------------------------------
// Kernel 2: implicit-GEMM conv via v_wmma_f32_16x16x32_bf16.
// Grid (H, T, B); block 256 = 8 waves; wave w owns o-tile [16w, 16w+16) and
// four 16-wide W tiles (four accumulators). Input row staged f32->bf16 in LDS.
// All four A-fragments are fetched as one ds_load clause before the four
// back-to-back WMMAs so LDS latency overlaps the matrix pipe.
// ---------------------------------------------------------------------------
__global__ __launch_bounds__(256) void conv_kernel(
    const float*  __restrict__ fmap,  // [B][C][T][H][W]
    const __bf16* __restrict__ wmod,  // [B][O][27][C]
    float*        __restrict__ out) { // [B][O][T][H][W]
  __shared__ __bf16 X[66 * XS];       // [wq 0..65][c 0..127 (+pad)]
  __shared__ float  OutS[Oo * OS];    // [o][w (+pad)]

  const int h = blockIdx.x;
  const int t = blockIdx.y;
  const int b = blockIdx.z;
  const int tid  = threadIdx.x;
  const int wave = tid >> 5;
  const int lane = tid & 31;
  const int hi   = lane >> 4;   // half-wave index
  const int mn   = lane & 15;   // M row (pixels) for A, N col (out ch) for B
  const int o0   = wave * 16;

  v8f acc0 = {}, acc1 = {}, acc2 = {}, acc3 = {};

  for (int kt = 0; kt < 3; ++kt) {
    const int tt = t + kt - 2;                       // causal pad (2,0)
    for (int kh = 0; kh < 3; ++kh) {
      const int hh = h + kh - 1;
      __syncthreads();  // previous-iteration readers done
      // ---- stage padded input row: X[wq][c] = bf16(fmap[b,c,tt,hh,wq-1]) ---
      for (int idx = tid; idx < 66 * Cc; idx += 256) {
        const int c  = idx / 66;
        const int wq = idx - c * 66;
        const int ww = wq - 1;
        float v = 0.f;
        if (tt >= 0 && (unsigned)hh < Hh && (unsigned)ww < Wn)
          v = fmap[(((size_t)(b * Cc + c) * Tt + tt) * Hh + hh) * Wn + ww];
        X[wq * XS + c] = (__bf16)v;
      }
      __syncthreads();

      // ---- GEMM over channel chunks (K=32) and kw taps -----------------
#pragma unroll
      for (int ic = 0; ic < 4; ++ic) {
#pragma unroll
        for (int kw = 0; kw < 3; ++kw) {
          const int kk = (kt * 3 + kh) * 3 + kw;
          // B fragment: elems 0..7: K=klo..klo+7, elems 8..15: K=klo+16..klo+23, klo = 8*hi
          const __bf16* bp = wmod +
              (((size_t)(b * Oo + o0 + mn) * KTAPS + kk) * Cc + ic * 32 + (hi << 3));
          v16bf bfrag;
          *(float4*)&bfrag       = *(const float4*)bp;
          *((float4*)&bfrag + 1) = *(const float4*)(bp + 16);

          // A fragments: 16-bit A layout -> elems 0..7: K=klo..klo+7,
          // elems 8..15: K=klo+16..klo+23, klo = 8*hi.
          // Load all four tiles first (one ds clause), then 4 WMMAs.
          const __bf16* ap0 = &X[( 0 + mn + kw) * XS + ic * 32 + (hi << 3)];
          const __bf16* ap1 = &X[(16 + mn + kw) * XS + ic * 32 + (hi << 3)];
          const __bf16* ap2 = &X[(32 + mn + kw) * XS + ic * 32 + (hi << 3)];
          const __bf16* ap3 = &X[(48 + mn + kw) * XS + ic * 32 + (hi << 3)];
          v16bf af0, af1, af2, af3;
          *(float4*)&af0       = *(const float4*)ap0;
          *((float4*)&af0 + 1) = *(const float4*)(ap0 + 16);
          *(float4*)&af1       = *(const float4*)ap1;
          *((float4*)&af1 + 1) = *(const float4*)(ap1 + 16);
          *(float4*)&af2       = *(const float4*)ap2;
          *((float4*)&af2 + 1) = *(const float4*)(ap2 + 16);
          *(float4*)&af3       = *(const float4*)ap3;
          *((float4*)&af3 + 1) = *(const float4*)(ap3 + 16);

          acc0 = __builtin_amdgcn_wmma_f32_16x16x32_bf16(
              false, af0, false, bfrag, (short)0, acc0, false, false);
          acc1 = __builtin_amdgcn_wmma_f32_16x16x32_bf16(
              false, af1, false, bfrag, (short)0, acc1, false, false);
          acc2 = __builtin_amdgcn_wmma_f32_16x16x32_bf16(
              false, af2, false, bfrag, (short)0, acc2, false, false);
          acc3 = __builtin_amdgcn_wmma_f32_16x16x32_bf16(
              false, af3, false, bfrag, (short)0, acc3, false, false);
        }
      }
    }
  }

  // ---- transpose through LDS so global stores are coalesced --------------
  __syncthreads();
#pragma unroll
  for (int wti = 0; wti < 4; ++wti) {
    const v8f a = (wti == 0) ? acc0 : (wti == 1) ? acc1 : (wti == 2) ? acc2 : acc3;
#pragma unroll
    for (int r = 0; r < 8; ++r) {
      const int m = r + (hi << 3);  // D layout: lanes16-31 hold M=r+8
      OutS[(o0 + mn) * OS + wti * 16 + m] = a[r];
    }
  }
  __syncthreads();
  for (int idx = tid; idx < Oo * Wn; idx += 256) {
    const int o = idx >> 6;
    const int w = idx & 63;
    out[(((size_t)(b * Oo + o) * Tt + t) * Hh + h) * Wn + w] = OutS[o * OS + w];
  }
}

extern "C" void kernel_launch(void* const* d_in, const int* in_sizes, int n_in,
                              void* d_out, int out_size, void* d_ws, size_t ws_size,
                              hipStream_t stream) {
  const float* fmap    = (const float*)d_in[0];  // [4][128][16][64][64]
  const float* cond    = (const float*)d_in[1];  // [4][128]
  const float* weights = (const float*)d_in[2];  // [128][128][3][3][3]
  float* out = (float*)d_out;
  __bf16* wmod = (__bf16*)d_ws;                  // 4*128*27*128 bf16 = 3.4 MB

  modw_kernel<<<Bn * Oo, 256, 0, stream>>>(weights, cond, wmod);
  conv_kernel<<<dim3(Hh, Tt, Bn), 256, 0, stream>>>(fmap, wmod, out);
}
